// DecoderInputEmbedding_5222680231968
// MI455X (gfx1250) — matrix operand order repaired and checked by reading, not compiled, not run on hardware
//
#include <hip/hip_runtime.h>
#include <hip/hip_bf16.h>
#include <math.h>

// ---------------------------------------------------------------------------
// Types for CDNA5 WMMA (wave32): v_wmma_f32_16x16x32_bf16
// ---------------------------------------------------------------------------
typedef __bf16 bf16;
typedef __attribute__((ext_vector_type(16))) __bf16 v16bf;
typedef __attribute__((ext_vector_type(8)))  __bf16 v8bf;
typedef __attribute__((ext_vector_type(8)))  float  v8f;

static __device__ __forceinline__ bf16 tobf(float f) { return (bf16)f; }

// A-fragment (16x32 bf16), per ISA 7.12.2:
//   lane m = lane&15 ; k = ((e>>3)<<4) + ((lane>>4)<<3) + (e&7)
// => two contiguous 8-element (16B) loads at k0+8*(lane>>4) and +16.
static __device__ __forceinline__ v16bf load_a(const bf16* base, int stride,
                                               int m0, int k0, int lane) {
  const bf16* p = base + (size_t)(m0 + (lane & 15)) * stride + k0 + ((lane >> 4) << 3);
  v8bf lo = *(const v8bf*)p;
  v8bf hi = *(const v8bf*)(p + 16);
  v16bf a;
#pragma unroll
  for (int i = 0; i < 8; ++i) { a[i] = lo[i]; a[i + 8] = hi[i]; }
  return a;
}

// B-fragment (32x16 bf16): lane holds col n = n0+(lane&15); k = ((e>>3)<<4) + ((lane>>4)<<3) + (e&7).
static __device__ __forceinline__ v16bf load_b(const bf16* base, int stride,
                                               int k0, int n0, int lane) {
  const bf16* p = base + (size_t)(k0 + ((lane >> 4) << 3)) * stride + n0 + (lane & 15);
  v16bf b;
#pragma unroll
  for (int i = 0; i < 8; ++i) {
    b[i] = p[(size_t)i * stride];
    b[i + 8] = p[(size_t)(i + 16) * stride];
  }
  return b;
}

static __device__ __forceinline__ v8f wmma_bf(v16bf a, v16bf b, v8f c) {
  return __builtin_amdgcn_wmma_f32_16x16x32_bf16(false, a, false, b, (short)0, c,
                                                 false, false);
}

// C/D layout: VGPR r -> row m0 + r + 8*(lane>=16); col n0 + (lane&15)

// ---------------------------------------------------------------------------
// Problem constants
// ---------------------------------------------------------------------------
#define SW   96
#define FB   64
#define EMB  512
#define NH   3
#define DH   32
#define BATCH 4
#define TLEN 1024
#define NSEQ (BATCH * TLEN)   // 4096
#define FDIM (SW * FB)        // 6144
#define NELEM (NSEQ * EMB)    // 2097152

// workspace offsets (bytes)
#define OFF_WQ     0u
#define OFF_WK     18432u
#define OFF_WV     36864u
#define OFF_ERT    55296u      // Er transposed: 32 x 64
#define OFF_W1     59392u
#define OFF_W2     133120u
#define OFF_WE     206848u
#define OFF_EMBT   6498304u    // (4096 x 6144) bf16
#define OFF_EMB2   56829952u   // (4096 x 512)  f32
#define OFF_STATS  65218560u   // 514 doubles
#define OFF_IDS    65223168u   // 4096 int
#define OFF_STARTS 65239552u   // 4096 int

// ---------------------------------------------------------------------------
// Kernel 1: convert all weights to bf16 (Er also transposed)
// ---------------------------------------------------------------------------
__global__ __launch_bounds__(256) void convert_weights(
    const float* Wq, const float* Wk, const float* Wv, const float* Er,
    const float* W1, const float* W2, const float* We,
    bf16* wq, bf16* wk, bf16* wv, bf16* ert, bf16* w1, bf16* w2, bf16* we) {
  int g = blockIdx.x * 256 + threadIdx.x;
  int stride = gridDim.x * 256;
  for (int i = g; i < FDIM * EMB; i += stride) we[i] = tobf(We[i]);
  for (int i = g; i < SW * SW; i += stride) {
    wq[i] = tobf(Wq[i]); wk[i] = tobf(Wk[i]); wv[i] = tobf(Wv[i]);
  }
  for (int i = g; i < SW * 4 * SW; i += stride) {
    w1[i] = tobf(W1[i]); w2[i] = tobf(W2[i]);
  }
  for (int i = g; i < DH * FB; i += stride) {   // ErT[d][m] = Er[m][d]
    int d = i >> 6, m = i & 63;
    ert[i] = tobf(Er[m * DH + d]);
  }
}

// ---------------------------------------------------------------------------
// Kernel 2: per-sequence relative attention + FFN, all WMMA bf16.
// One workgroup (128 thr = 4 waves) per sequence of L=64, D=96.
// Dynamic LDS layout (100 KB), phase-overlapped:
//   [0      ,12288) sX  64x96 bf16      (phase 1)   |  sH 64x384 bf16 (phase 3)
//   [12288  ,24576) sQ  64x96 bf16                  |  (overlapped by sH)
//   [24576  ,36864) sKT 96x64 bf16 (K transposed)   |  (overlapped by sH)
//   [36864  ,49152) sV  64x96 bf16                  |  (overlapped by sH)
//   [49152  ,65536) sQER 64x64 f32
//   [65536  ,81920) sS   64x64 f32
//   [81920  ,90112) sP   64x64 bf16
//   [90112  ,102400) sO  64x96 bf16
// ---------------------------------------------------------------------------
#define ATTN_LDS 102400
__global__ __launch_bounds__(128) void attn_ffn(
    const float* x, const float* bq, const float* bk, const float* bv,
    const float* b1, const float* b2,
    const bf16* Wq, const bf16* Wk, const bf16* Wv, const bf16* ErT,
    const bf16* W1, const bf16* W2, bf16* emb_bt) {
  extern __shared__ __align__(128) char smem[];
  bf16*  sX   = (bf16*)smem;
  bf16*  sQ   = (bf16*)(smem + 12288);
  bf16*  sKT  = (bf16*)(smem + 24576);
  bf16*  sV   = (bf16*)(smem + 36864);
  float* sQER = (float*)(smem + 49152);
  float* sS   = (float*)(smem + 65536);
  bf16*  sP   = (bf16*)(smem + 81920);
  bf16*  sO   = (bf16*)(smem + 90112);
  bf16*  sH   = (bf16*)smem;            // 64x384, reuses phase-1 space

  const int tid  = threadIdx.x;
  const int wave = tid >> 5;
  const int lane = tid & 31;
  const int nseq = blockIdx.x;
  const float* xrow = x + (size_t)nseq * FDIM;

  // Stage X[l][d] = x[n, d*64+l] into LDS as bf16 (coalesced global reads).
  for (int i = tid; i < FDIM; i += 128) {
    int d = i >> 6, l = i & 63;
    sX[l * SW + d] = tobf(xrow[i]);
  }
  __syncthreads();

  // ---- Q/K/V = X*W + b  (72 output tiles of 16x16, K=96 -> 3 wmma steps) ----
  for (int t = wave; t < 72; t += 4) {
    int which = t / 24, rem = t % 24, mt = rem / 6, nt = rem % 6;
    const bf16* W = (which == 0) ? Wq : (which == 1) ? Wk : Wv;
    const float* bias = (which == 0) ? bq : (which == 1) ? bk : bv;
    v8f c = {};
#pragma unroll
    for (int ks = 0; ks < 3; ++ks) {
      v16bf a = load_a(sX, SW, mt * 16, ks * 32, lane);
      v16bf b = load_b(W,  SW, ks * 32, nt * 16, lane);
      c = wmma_bf(a, b, c);
    }
    int n  = nt * 16 + (lane & 15);
    int mb = mt * 16 + ((lane >> 4) << 3);
    float bz = bias[n];
    if (which == 1) {                      // store K transposed: KT[d][l]
#pragma unroll
      for (int r = 0; r < 8; ++r) sKT[n * 64 + mb + r] = tobf(c[r] + bz);
    } else {
      bf16* dst = (which == 0) ? sQ : sV;
#pragma unroll
      for (int r = 0; r < 8; ++r) dst[(mb + r) * SW + n] = tobf(c[r] + bz);
    }
  }
  __syncthreads();

  // ---- per-head attention (heads sequential; buffers reused) ----
  for (int h = 0; h < NH; ++h) {
    // qer[l][m] = Qh[l][:] . Er[m][:]  -> B = ErT (32x64), one k-step
    for (int t = wave; t < 16; t += 4) {
      int mt = t >> 2, nt = t & 3;
      v16bf a = load_a(sQ, SW, mt * 16, h * DH, lane);
      v16bf b = load_b(ErT, 64, 0, nt * 16, lane);
      v8f c = {};
      c = wmma_bf(a, b, c);
      int n = nt * 16 + (lane & 15), mb = mt * 16 + ((lane >> 4) << 3);
#pragma unroll
      for (int r = 0; r < 8; ++r) sQER[(mb + r) * 64 + n] = c[r];
    }
    __syncthreads();
    // scores = (Q Kh^T + skew(qer)) / sqrt(dh) with causal mask
    for (int t = wave; t < 16; t += 4) {
      int mt = t >> 2, nt = t & 3;
      v16bf a = load_a(sQ, SW, mt * 16, h * DH, lane);
      v16bf b = load_b(sKT, 64, h * DH, nt * 16, lane);
      v8f c = {};
      c = wmma_bf(a, b, c);
      int n = nt * 16 + (lane & 15), mb = mt * 16 + ((lane >> 4) << 3);
#pragma unroll
      for (int r = 0; r < 8; ++r) {
        int m = mb + r;
        float s = (n > m) ? -3.0e38f
                          : (c[r] + sQER[m * 64 + (n - m + 63)]) * 0.17677669529663687f;
        sS[m * 64 + n] = s;
      }
    }
    __syncthreads();
    // softmax per row (64 rows handled by threads 0..63)
    if (tid < 64) {
      const float* row = sS + tid * 64;
      float mx = -3.4e38f;
      for (int j = 0; j < 64; ++j) mx = fmaxf(mx, row[j]);
      float sum = 0.f;
      for (int j = 0; j < 64; ++j) sum += __expf(row[j] - mx);
      float inv = 1.0f / sum;
      for (int j = 0; j < 64; ++j)
        sP[tid * 64 + j] = tobf(__expf(row[j] - mx) * inv);
    }
    __syncthreads();
    // out_h = P * Vh  (64x32, K=64 -> 2 steps)
    for (int t = wave; t < 8; t += 4) {
      int mt = t >> 1, nt = t & 1;
      v8f c = {};
#pragma unroll
      for (int ks = 0; ks < 2; ++ks) {
        v16bf a = load_a(sP, 64, mt * 16, ks * 32, lane);
        v16bf b = load_b(sV, SW, ks * 32, h * DH + nt * 16, lane);
        c = wmma_bf(a, b, c);
      }
      int n = h * DH + nt * 16 + (lane & 15), mb = mt * 16 + ((lane >> 4) << 3);
#pragma unroll
      for (int r = 0; r < 8; ++r) sO[(mb + r) * SW + n] = tobf(c[r]);
    }
    __syncthreads();
  }

  // ---- FFN1: H = relu(O*W1 + b1) (64x384, K=96) ----
  for (int t = wave; t < 96; t += 4) {
    int mt = t / 24, nt = t % 24;
    v8f c = {};
#pragma unroll
    for (int ks = 0; ks < 3; ++ks) {
      v16bf a = load_a(sO, SW, mt * 16, ks * 32, lane);
      v16bf b = load_b(W1, 4 * SW, ks * 32, nt * 16, lane);
      c = wmma_bf(a, b, c);
    }
    int n = nt * 16 + (lane & 15), mb = mt * 16 + ((lane >> 4) << 3);
    float bz = b1[n];
#pragma unroll
    for (int r = 0; r < 8; ++r)
      sH[(mb + r) * (4 * SW) + n] = tobf(fmaxf(c[r] + bz, 0.0f));
  }
  __syncthreads();

  // ---- FFN2: Y = H*W2 + b2 (64x96, K=384), store to emb_bt[n][d*64+l] ----
  bf16* orow = emb_bt + (size_t)nseq * FDIM;
  for (int t = wave; t < 24; t += 4) {
    int mt = t / 6, nt = t % 6;
    v8f c = {};
#pragma unroll
    for (int ks = 0; ks < 12; ++ks) {
      v16bf a = load_a(sH, 4 * SW, mt * 16, ks * 32, lane);
      v16bf b = load_b(W2, SW, ks * 32, nt * 16, lane);
      c = wmma_bf(a, b, c);
    }
    int n = nt * 16 + (lane & 15), mb = mt * 16 + ((lane >> 4) << 3);
    float bz = b2[n];
    v8bf o;
#pragma unroll
    for (int r = 0; r < 8; ++r) o[r] = tobf(c[r] + bz);
    *(v8bf*)(orow + n * 64 + mb) = o;   // contiguous 16B store (rows l)
  }
}

// ---------------------------------------------------------------------------
// Kernel 3: emb2 = emb_bt(4096x6144) @ We(6144x512) + be, f32 accum.
// WG = 8 waves (256 thr), 128x64 output tile.
// A chunks (128x64 bf16 = 16 KB) double-buffered in LDS via the CDNA5 async
// LDS path: global_load_async_to_lds_b128 (ASYNCcnt) + s_wait_asynccnt.
// Wave w: m-half = w>>2 (64 rows), n-strip = w&3 (16 cols), 4 m-tiles each.
// ---------------------------------------------------------------------------
#define G2_LDS 32768
__global__ __launch_bounds__(256) void gemm2(const bf16* A, const bf16* Bw,
                                             const float* be, float* C) {
  extern __shared__ __align__(128) char g2smem[];
  bf16* sA = (bf16*)g2smem;                 // 2 buffers x (128x64) bf16
  const int tid  = threadIdx.x;
  const int wave = tid >> 5, lane = tid & 31;
  const int mg = wave >> 2;                 // 0..1
  const int ns = wave & 3;                  // 0..3
  const int m0 = blockIdx.x * 128;
  const int n0 = blockIdx.y * 64 + ns * 16;
  const unsigned ldsbase = __builtin_amdgcn_groupstaticsize();

  // issue one 128x64 chunk: 1024 x 16B segments / 256 thr = 4 per thread
  // seg -> row = seg>>3, col = (seg&7)*8
  auto issue_chunk = [&](int kc, int buf) {
#pragma unroll
    for (int j = 0; j < 4; ++j) {
      int seg = tid + 256 * j;
      int row = seg >> 3, col = (seg & 7) * 8;
      const bf16* g = A + (size_t)(m0 + row) * FDIM + kc * 64 + col;
      unsigned l = ldsbase + (unsigned)(buf * 16384 + (row * 64 + col) * 2);
      asm volatile("global_load_async_to_lds_b128 %0, %1, off"
                   :: "v"(l), "v"(g) : "memory");
    }
  };

  v8f acc0 = {}, acc1 = {}, acc2 = {}, acc3 = {};
  const int NC = FDIM / 64;                 // 96 chunks
  issue_chunk(0, 0);
  for (int c = 0; c < NC; ++c) {
    int buf = c & 1;
    if (c + 1 < NC) {
      issue_chunk(c + 1, buf ^ 1);
      asm volatile("s_wait_asynccnt 0x4" ::: "memory");   // chunk c landed
    } else {
      asm volatile("s_wait_asynccnt 0x0" ::: "memory");
    }
    __syncthreads();                        // all waves' segments visible
    const bf16* a_base = sA + buf * 8192 + mg * 4096;     // 64x64, stride 64
#pragma unroll
    for (int kk = 0; kk < 2; ++kk) {
      v16bf b = load_b(Bw, EMB, c * 64 + kk * 32, n0, lane);
      acc0 = wmma_bf(load_a(a_base, 64,  0, kk * 32, lane), b, acc0);
      acc1 = wmma_bf(load_a(a_base, 64, 16, kk * 32, lane), b, acc1);
      acc2 = wmma_bf(load_a(a_base, 64, 32, kk * 32, lane), b, acc2);
      acc3 = wmma_bf(load_a(a_base, 64, 48, kk * 32, lane), b, acc3);
    }
    __syncthreads();                        // done reading buf before overwrite
  }

  int n = n0 + (lane & 15);
  float bz = be[n];
  v8f accs[4] = {acc0, acc1, acc2, acc3};
#pragma unroll
  for (int i = 0; i < 4; ++i) {
    int mb = m0 + mg * 64 + i * 16 + ((lane >> 4) << 3);
#pragma unroll
    for (int r = 0; r < 8; ++r) C[(size_t)(mb + r) * EMB + n] = accs[i][r] + bz;
  }
}

// ---------------------------------------------------------------------------
// Kernels 4/5: deterministic two-stage sum / sumsq reduction (double).
// stats: [0..255]=partial sums, [256..511]=partial sumsq, [512]=mu, [513]=rstd
// ---------------------------------------------------------------------------
__global__ __launch_bounds__(256) void reduce_partial(const float* e, double* stats) {
  __shared__ double ss[256], sq[256];
  int tid = threadIdx.x;
  double s = 0.0, q = 0.0;
  for (int i = blockIdx.x * 256 + tid; i < NELEM; i += 256 * 256) {
    double v = (double)e[i];
    s += v; q += v * v;
  }
  ss[tid] = s; sq[tid] = q;
  __syncthreads();
  for (int o = 128; o > 0; o >>= 1) {
    if (tid < o) { ss[tid] += ss[tid + o]; sq[tid] += sq[tid + o]; }
    __syncthreads();
  }
  if (tid == 0) { stats[blockIdx.x] = ss[0]; stats[256 + blockIdx.x] = sq[0]; }
}

__global__ __launch_bounds__(256) void stats_final(double* stats) {
  __shared__ double ss[256], sq[256];
  int tid = threadIdx.x;
  ss[tid] = stats[tid]; sq[tid] = stats[256 + tid];
  __syncthreads();
  for (int o = 128; o > 0; o >>= 1) {
    if (tid < o) { ss[tid] += ss[tid + o]; sq[tid] += sq[tid + o]; }
    __syncthreads();
  }
  if (tid == 0) {
    double n = (double)NELEM;
    double mu = ss[0] / n;
    double var = sq[0] / n - mu * mu;
    stats[512] = mu;
    stats[513] = 1.0 / sqrt(var + 1e-8);
  }
}

// ---------------------------------------------------------------------------
// Kernel 6: block ids (inclusive cumsum of o_enc minus first) + start flags.
// ---------------------------------------------------------------------------
__global__ void scan_blocks(const int* o_enc, int* ids, int* starts) {
  int b = blockIdx.x;
  if (threadIdx.x == 0) {
    const int* o = o_enc + b * TLEN;
    int* id = ids + b * TLEN;
    int* st = starts + b * TLEN;
    int c = 0, first = o[0];
    for (int t = 0; t < TLEN; ++t) { c += o[t]; id[t] = c - first; }
    st[0] = 1;
    for (int t = 1; t < TLEN; ++t) st[t] = (id[t] != id[t - 1]) ? 1 : 0;
  }
}

// ---------------------------------------------------------------------------
// Kernel 7: out = normalize(emb2) + r_enc + (is_start ? block mean : 0)
// One workgroup per token (blocks are contiguous in t).
// ---------------------------------------------------------------------------
__global__ __launch_bounds__(256) void finalize(const float* emb2, const float* r_enc,
                                                const int* ids, const int* starts,
                                                const double* stats, float* out) {
  int tok = blockIdx.x;
  int b = tok >> 10, t = tok & (TLEN - 1);
  float mu = (float)stats[512], rstd = (float)stats[513];
  int myid = ids[tok];
  int isst = starts[tok];
  int tend = t + 1;
  if (isst) { while (tend < TLEN && ids[b * TLEN + tend] == myid) ++tend; }
  float invcnt = 1.0f / (float)(tend - t);
  for (int e = threadIdx.x; e < EMB; e += 256) {
    float v = (emb2[(size_t)tok * EMB + e] - mu) * rstd;
    float o = v + r_enc[(size_t)tok * EMB + e];
    if (isst) {
      float s = 0.f;
      for (int tt = t; tt < tend; ++tt)
        s += (emb2[(size_t)(b * TLEN + tt) * EMB + e] - mu) * rstd;
      o += s * invcnt;
    }
    out[(size_t)tok * EMB + e] = o;
  }
}

// ---------------------------------------------------------------------------
extern "C" void kernel_launch(void* const* d_in, const int* in_sizes, int n_in,
                              void* d_out, int out_size, void* d_ws, size_t ws_size,
                              hipStream_t stream) {
  (void)in_sizes; (void)n_in; (void)out_size; (void)ws_size;
  const float* x     = (const float*)d_in[0];
  const int*   o_enc = (const int*)  d_in[1];
  const float* r_enc = (const float*)d_in[2];
  const float* Wq    = (const float*)d_in[3];
  const float* bq    = (const float*)d_in[4];
  const float* Wk    = (const float*)d_in[5];
  const float* bk    = (const float*)d_in[6];
  const float* Wv    = (const float*)d_in[7];
  const float* bv    = (const float*)d_in[8];
  const float* Er    = (const float*)d_in[9];
  const float* W1    = (const float*)d_in[10];
  const float* b1    = (const float*)d_in[11];
  const float* W2    = (const float*)d_in[12];
  const float* b2    = (const float*)d_in[13];
  const float* We    = (const float*)d_in[14];
  const float* be    = (const float*)d_in[15];
  float* out = (float*)d_out;
  char* ws = (char*)d_ws;

  bf16*  wq_bf  = (bf16*)(ws + OFF_WQ);
  bf16*  wk_bf  = (bf16*)(ws + OFF_WK);
  bf16*  wv_bf  = (bf16*)(ws + OFF_WV);
  bf16*  ert_bf = (bf16*)(ws + OFF_ERT);
  bf16*  w1_bf  = (bf16*)(ws + OFF_W1);
  bf16*  w2_bf  = (bf16*)(ws + OFF_W2);
  bf16*  we_bf  = (bf16*)(ws + OFF_WE);
  bf16*  emb_bt = (bf16*)(ws + OFF_EMBT);
  float* emb2   = (float*)(ws + OFF_EMB2);
  double* stats = (double*)(ws + OFF_STATS);
  int*   ids    = (int*)(ws + OFF_IDS);
  int*   starts = (int*)(ws + OFF_STARTS);

  convert_weights<<<1024, 256, 0, stream>>>(Wq, Wk, Wv, Er, W1, W2, We,
                                            wq_bf, wk_bf, wv_bf, ert_bf,
                                            w1_bf, w2_bf, we_bf);
  attn_ffn<<<NSEQ, 128, ATTN_LDS, stream>>>(x, bq, bk, bv, b1, b2,
                                            wq_bf, wk_bf, wv_bf, ert_bf,
                                            w1_bf, w2_bf, emb_bt);
  gemm2<<<dim3(NSEQ / 128, EMB / 64), 256, G2_LDS, stream>>>(emb_bt, we_bf, be, emb2);
  reduce_partial<<<256, 256, 0, stream>>>(emb2, stats);
  stats_final<<<1, 256, 0, stream>>>(stats);
  scan_blocks<<<BATCH, 32, 0, stream>>>(o_enc, ids, starts);
  finalize<<<NSEQ, 256, 0, stream>>>(emb2, r_enc, ids, starts, stats, out);
}
